// XDeepFMLayer_8211977470670
// MI455X (gfx1250) — matrix operand order repaired and checked by reading, not compiled, not run on hardware
//
#include <hip/hip_runtime.h>
#include <hip/hip_bf16.h>

// ---- problem constants ----
#define BB 4096
#define FF 39
#define DD 16
#define IN_DIM 624       // F*D
#define H1C 128
#define H2C 128
#define K1V 1521         // 39*39
#define K1P 1536         // padded to 64
#define K2V 2496         // 64*39 (mult of 64: 2496 = 39*64)
#define DEEP1 400
#define DEEP2 400
#define KD1 640          // 624 padded
#define KD2 416          // 400 padded
#define BN_EPS 1e-5f

typedef unsigned int   u32;
typedef unsigned short u16;

typedef __attribute__((ext_vector_type(16))) __bf16 v16bf;
typedef __attribute__((ext_vector_type(8)))  float  v8f;

__device__ __forceinline__ u16 f2bf(float f) {
    union { float f; u32 u; } c; c.f = f;
    u32 u = c.u + 0x7FFFu + ((c.u >> 16) & 1u);   // round-to-nearest-even
    return (u16)(u >> 16);
}

union BFrag { v16bf v; u32 u[8]; };

// A-fragment: 16x32 bf16, row-major source (row m, stride ld elements).
// Layout: lanes 0-15 (M=l) K 0..15; lanes 16-31 K 16..31 (matches B fragment).
__device__ __forceinline__ void load_a_frag(BFrag& fr, const u16* base, int ld) {
    int lane = threadIdx.x & 31;
    const u16* row = base + (size_t)(lane & 15) * ld;
    int kb = (lane >> 4) ? 16 : 0;
#pragma unroll
    for (int i = 0; i < 8; ++i) {
        int k0 = kb + 2 * i;
        fr.u[i] = *(const u32*)(row + k0);
    }
}

// B-fragment: 32x16 bf16 where B[k][n] = Wt[n][k] (transposed source, k contiguous).
// lanes 0-15 (N=l) hold K=0..15, lanes 16-31 hold K=16..31, 2 K per VGPR.
__device__ __forceinline__ void load_b_frag(BFrag& fr, const u16* base, int ld) {
    int lane = threadIdx.x & 31;
    const u16* row = base + (size_t)(lane & 15) * ld;
    int kb = (lane >> 4) ? 16 : 0;
#pragma unroll
    for (int i = 0; i < 8; ++i)
        fr.u[i] = *(const u32*)(row + kb + 2 * i);
}

__device__ __forceinline__ v8f wmma_bf16(const BFrag& a, const BFrag& b, v8f c) {
    return __builtin_amdgcn_wmma_f32_16x16x32_bf16(false, a.v, false, b.v,
                                                   (short)0, c, false, false);
}

// ---------------- kernel 0: weight prep (f32 -> bf16, transpose/pad) ----------------
__global__ void prep_kernel(const float* __restrict__ cw1, const float* __restrict__ cw2,
                            const float* __restrict__ dw1, const float* __restrict__ dw2,
                            u16* __restrict__ W1bf, u16* __restrict__ W2bf,
                            u16* __restrict__ W1dT, u16* __restrict__ W2dT) {
    int i = blockIdx.x * 256 + threadIdx.x;
    const int N1 = H1C * K1P, N2 = H2C * K2V, N3 = DEEP1 * KD1, N4 = DEEP2 * KD2;
    if (i < N1) {                       // conv_w1 (128,1521) -> [128][1536]
        int o = i / K1P, k = i % K1P;
        W1bf[i] = f2bf(k < K1V ? cw1[o * K1V + k] : 0.f);
        return;
    }
    i -= N1;
    if (i < N2) { W2bf[i] = f2bf(cw2[i]); return; }   // (128,2496) same layout
    i -= N2;
    if (i < N3) {                       // deep_w1 (624,400) -> Wt [400][640]
        int n = i / KD1, k = i % KD1;
        W1dT[i] = f2bf(k < IN_DIM ? dw1[(size_t)k * DEEP1 + n] : 0.f);
        return;
    }
    i -= N3;
    if (i < N4) {                       // deep_w2 (400,400) -> Wt [400][416]
        int n = i / KD2, k = i % KD2;
        W2dT[i] = f2bf(k < DEEP1 ? dw2[(size_t)k * DEEP2 + n] : 0.f);
    }
}

// ---------------- kernel 1: embedding gather ----------------
// flatf [B][624] f32 ; flatbf [B][640] bf16 (padded) ; x0row [B*16][40] f32 (r=b*16+d)
__global__ void gather_kernel(const int* __restrict__ idx, const float* __restrict__ emb,
                              float* __restrict__ flatf, u16* __restrict__ flatbf,
                              float* __restrict__ x0row) {
    int i = blockIdx.x * 256 + threadIdx.x;      // B*(F+1)
    int b = i / (FF + 1), f = i % (FF + 1);
    if (b >= BB) return;
    if (f < FF) {
        const float* src = emb + (size_t)idx[b * FF + f] * DD;
#pragma unroll
        for (int d = 0; d < DD; ++d) {
            float v = src[d];
            flatf[(size_t)b * IN_DIM + f * DD + d] = v;
            flatbf[(size_t)b * KD1 + f * DD + d] = f2bf(v);
            x0row[(size_t)(b * DD + d) * 40 + f] = v;
        }
    } else {
        for (int d = 0; d < DD; ++d) flatbf[(size_t)b * KD1 + IN_DIM + d] = 0;  // pad
    }
}

// ---------------- CIN layer (fused outer-product GEMM via WMMA) ----------------
// One workgroup per b (16 rows = all d of one sample). 8 waves = 128 output channels.
// 64-wide K iterations, double-buffered LDS staging, 2 WMMAs/wave/iter, 1 barrier/iter.
__global__ __launch_bounds__(256)
void cin_kernel(const float* __restrict__ x0row, const float* __restrict__ prevrow,
                int ld_prev, int hprev,
                const u16* __restrict__ Wbf, int ldw, int kvalid, int niter,
                const float* __restrict__ bias, float* __restrict__ nxtrow,
                float* __restrict__ res, int split) {
    __shared__ float x0s[16][40];
    __shared__ float prevs[16][64];
    __shared__ u16   zs[2][16][64];
    __shared__ u16   wls[2][128][64];
    __shared__ float outs[128][16];

    int b = blockIdx.x;
    int tid = threadIdx.x;
    int wave = tid >> 5;

    for (int i = tid; i < 16 * FF; i += 256) {
        int d = i / FF, m = i % FF;
        x0s[d][m] = x0row[(size_t)(b * 16 + d) * 40 + m];
    }
    for (int i = tid; i < 16 * hprev; i += 256) {
        int d = i / hprev, h = i % hprev;
        prevs[d][h] = prevrow[(size_t)(b * 16 + d) * ld_prev + h];
    }

    // z-build mapping: thread -> (d, 4 consecutive kk); incremental (h,m) per chunk.
    int dz  = tid >> 4;
    int kkb = (tid & 15) * 4;
    int h = kkb / 39;              // one division at init only
    int m = kkb - h * 39;
    // W staging mapping: thread loads 16 ushorts of one row per iter.
    int wrow = tid >> 1, wcol = (tid & 1) * 32;

    __syncthreads();

    v8f acc = {};
    for (int it = 0; it < niter; ++it) {
        int p = it & 1;
        int kb = it * 64;
        // ---- stage z chunk: z[d][kk] = prev[d][h]*x0[d][m], branchless tail mask ----
        {
            int hh = h, mm = m;
            u32 pk[2];
#pragma unroll
            for (int jj = 0; jj < 2; ++jj) {
                u32 lo, hi;
#pragma unroll
                for (int e = 0; e < 2; ++e) {
                    int j = jj * 2 + e;
                    float zz = prevs[dz][hh] * x0s[dz][mm];
                    zz = (kb + kkb + j < kvalid) ? zz : 0.f;   // v_cndmask, no branch
                    u32 bf = f2bf(zz);
                    if (e == 0) lo = bf; else hi = bf;
                    ++mm; if (mm >= 39) { mm = 0; ++hh; }
                }
                pk[jj] = lo | (hi << 16);
            }
            *(u32*)&zs[p][dz][kkb]     = pk[0];
            *(u32*)&zs[p][dz][kkb + 2] = pk[1];
            // advance (h,m) by 64 = 39 + 25 (guaranteed >= 1 wrap, <= 2)
            m += 25; ++h;
            if (m >= 39) { m -= 39; ++h; }
        }
        // ---- stage W chunk: wls[o][c] = W[o][kb+c] for c in 0..15, 32..47 (8KB from L2) ----
        {
            const u16* src = Wbf + (size_t)wrow * ldw + kb + wcol;
            u32 tmp[8];
#pragma unroll
            for (int j = 0; j < 8; ++j) tmp[j] = *(const u32*)(src + 2 * j);
#pragma unroll
            for (int j = 0; j < 8; ++j) *(u32*)&wls[p][wrow][wcol + 2 * j] = tmp[j];
        }
        __syncthreads();
        BFrag a0, a1, w0, w1;
        load_a_frag(a0, &zs[p][0][0], 64);
        load_a_frag(a1, &zs[p][0][32], 64);
        load_b_frag(w0, &wls[p][wave * 16][0], 64);
        load_b_frag(w1, &wls[p][wave * 16][32], 64);
        acc = wmma_bf16(a0, w0, acc);
        acc = wmma_bf16(a1, w1, acc);
    }
    __syncthreads();
    // D tile -> LDS: M=d rows, N=o cols
    {
        int lane = tid & 31;
        int n = lane & 15, mo = (lane >> 4) * 8;
#pragma unroll
        for (int v = 0; v < 8; ++v) outs[wave * 16 + n][v + mo] = acc[v];
    }
    __syncthreads();
    if (tid < 128) {
        int o = tid;
        float bv = bias[o], s = 0.f;
#pragma unroll
        for (int d = 0; d < 16; ++d) {
            float v = outs[o][d] + bv;
            v = v > 0.f ? v : 0.f;                      // relu
            if (split && o < 64) nxtrow[(size_t)(b * 16 + d) * 64 + o] = v;
            s += v;                                     // sum over axis d
        }
        if (split) { if (o >= 64) res[b * 192 + (o - 64)] = s; }
        else res[b * 192 + 64 + o] = s;
    }
}

// ---------------- generic bf16 GEMM: C[M][N] = A[M][Kp] * Bt[N][Kp]^T + bias ----------------
__global__ __launch_bounds__(256)
void gemm_bf16_kernel(const u16* __restrict__ A, int lda, const u16* __restrict__ Bt, int ldb,
                      const float* __restrict__ bias, float* __restrict__ C,
                      int M, int N, int Kpad) {
    int w = blockIdx.x * 8 + (threadIdx.x >> 5);   // wave-per-16x16-tile
    int tilesN = N >> 4;
    int tm = w / tilesN, tn = w % tilesN;
    if (tm * 16 >= M) return;                       // wave-uniform
    const u16* ab = A + (size_t)(tm * 16) * lda;
    const u16* bb = Bt + (size_t)(tn * 16) * ldb;
    v8f acc = {};
    for (int kb = 0; kb < Kpad; kb += 32) {
        BFrag a, b;
        load_a_frag(a, ab + kb, lda);
        load_b_frag(b, bb + kb, ldb);
        acc = wmma_bf16(a, b, acc);
    }
    int lane = threadIdx.x & 31;
    int n = lane & 15, mo = (lane >> 4) * 8;
    int gn = tn * 16 + n;
    float bv = bias ? bias[gn] : 0.f;
#pragma unroll
    for (int v = 0; v < 8; ++v) {
        int gm = tm * 16 + v + mo;
        C[(size_t)gm * N + gn] = acc[v] + bv;
    }
}

// ---------------- BN stats: biased batch mean/var per column ----------------
__global__ __launch_bounds__(256)
void bnstats_kernel(const float* __restrict__ X, int N, float* __restrict__ mean,
                    float* __restrict__ rstd) {
    __shared__ float s1[256], s2[256];
    int j = blockIdx.x;
    float a = 0.f, q = 0.f;
    for (int i = threadIdx.x; i < BB; i += 256) {
        float x = X[(size_t)i * N + j];
        a += x; q += x * x;
    }
    s1[threadIdx.x] = a; s2[threadIdx.x] = q;
    __syncthreads();
    for (int s = 128; s > 0; s >>= 1) {
        if (threadIdx.x < s) { s1[threadIdx.x] += s1[threadIdx.x + s];
                               s2[threadIdx.x] += s2[threadIdx.x + s]; }
        __syncthreads();
    }
    if (threadIdx.x == 0) {
        float mm = s1[0] / BB;
        float v = s2[0] / BB - mm * mm;
        mean[j] = mm;
        rstd[j] = rsqrtf(v + BN_EPS);
    }
}

// BN + relu -> bf16 (padded to KD2 cols) for next GEMM
__global__ void bnapply_bf16_kernel(const float* __restrict__ X, const float* __restrict__ mean,
                                    const float* __restrict__ rstd, const float* __restrict__ g,
                                    const float* __restrict__ be, u16* __restrict__ Y) {
    int i = blockIdx.x * 256 + threadIdx.x;   // B*KD2
    int r = i / KD2, j = i % KD2;
    if (r >= BB) return;
    u16 o = 0;
    if (j < DEEP1) {
        float x = X[(size_t)r * DEEP1 + j];
        float y = g[j] * (x - mean[j]) * rstd[j] + be[j];
        o = f2bf(y > 0.f ? y : 0.f);
    }
    Y[i] = o;
}

// BN + relu -> f32
__global__ void bnapply_f32_kernel(const float* __restrict__ X, const float* __restrict__ mean,
                                   const float* __restrict__ rstd, const float* __restrict__ g,
                                   const float* __restrict__ be, float* __restrict__ Y) {
    int i = blockIdx.x * 256 + threadIdx.x;   // B*DEEP2
    if (i >= BB * DEEP2) return;
    int j = i % DEEP2;
    float y = g[j] * (X[i] - mean[j]) * rstd[j] + be[j];
    Y[i] = y > 0.f ? y : 0.f;
}

// ---------------- final: linear part + concat dot out_w ----------------
__global__ void final_kernel(const float* __restrict__ flatf, const float* __restrict__ lw,
                             const float* __restrict__ lb, const float* __restrict__ res,
                             const float* __restrict__ y2, const float* __restrict__ ow,
                             const float* __restrict__ ob, float* __restrict__ out) {
    int b = blockIdx.x * 256 + threadIdx.x;
    if (b >= BB) return;
    float lin = lb[0];
    for (int k = 0; k < IN_DIM; ++k) lin += flatf[(size_t)b * IN_DIM + k] * lw[k];
    float s = ob[0] + lin * ow[0];
    for (int j = 0; j < 192; ++j) s += res[b * 192 + j] * ow[1 + j];
    for (int j = 0; j < DEEP2; ++j) s += y2[(size_t)b * DEEP2 + j] * ow[193 + j];
    out[b] = s;
}

extern "C" void kernel_launch(void* const* d_in, const int* in_sizes, int n_in,
                              void* d_out, int out_size, void* d_ws, size_t ws_size,
                              hipStream_t stream) {
    (void)in_sizes; (void)n_in; (void)out_size; (void)ws_size;
    const int*   feat_index = (const int*)  d_in[0];
    const float* emb_table  = (const float*)d_in[2];
    const float* linear_w   = (const float*)d_in[3];
    const float* linear_b   = (const float*)d_in[4];
    const float* conv_w1    = (const float*)d_in[5];
    const float* conv_b1    = (const float*)d_in[6];
    const float* conv_w2    = (const float*)d_in[7];
    const float* conv_b2    = (const float*)d_in[8];
    const float* deep_w1    = (const float*)d_in[9];
    const float* deep_b1    = (const float*)d_in[10];
    const float* bn_g1      = (const float*)d_in[11];
    const float* bn_b1      = (const float*)d_in[12];
    const float* deep_w2    = (const float*)d_in[13];
    const float* deep_b2    = (const float*)d_in[14];
    const float* bn_g2      = (const float*)d_in[15];
    const float* bn_b2      = (const float*)d_in[16];
    const float* out_w      = (const float*)d_in[17];
    const float* out_b      = (const float*)d_in[18];
    float* out = (float*)d_out;

    char* w = (char*)d_ws;
    size_t off = 0;
    auto alloc = [&](size_t bytes) -> void* {
        void* p = w + off;
        off += (bytes + 255) & ~(size_t)255;
        return p;
    };
    float* flatf  = (float*)alloc((size_t)BB * IN_DIM * 4);
    u16*   flatbf = (u16*)  alloc((size_t)BB * KD1 * 2);
    float* x0row  = (float*)alloc((size_t)BB * 16 * 40 * 4);
    u16*   W1bf   = (u16*)  alloc((size_t)H1C * K1P * 2);
    u16*   W2bf   = (u16*)  alloc((size_t)H2C * K2V * 2);
    u16*   W1dT   = (u16*)  alloc((size_t)DEEP1 * KD1 * 2);
    u16*   W2dT   = (u16*)  alloc((size_t)DEEP2 * KD2 * 2);
    float* nxtrow = (float*)alloc((size_t)BB * 16 * 64 * 4);
    float* res    = (float*)alloc((size_t)BB * 192 * 4);
    float* y1raw  = (float*)alloc((size_t)BB * DEEP1 * 4);
    u16*   y1nbf  = (u16*)  alloc((size_t)BB * KD2 * 2);
    float* y2raw  = (float*)alloc((size_t)BB * DEEP2 * 4);
    float* y2     = (float*)alloc((size_t)BB * DEEP2 * 4);
    float* mean1  = (float*)alloc(DEEP1 * 4);
    float* rstd1  = (float*)alloc(DEEP1 * 4);
    float* mean2  = (float*)alloc(DEEP2 * 4);
    float* rstd2  = (float*)alloc(DEEP2 * 4);

    const int prepN = H1C * K1P + H2C * K2V + DEEP1 * KD1 + DEEP2 * KD2;  // 938496
    prep_kernel<<<(prepN + 255) / 256, 256, 0, stream>>>(
        conv_w1, conv_w2, deep_w1, deep_w2, W1bf, W2bf, W1dT, W2dT);

    gather_kernel<<<(BB * (FF + 1) + 255) / 256, 256, 0, stream>>>(
        feat_index, emb_table, flatf, flatbf, x0row);

    // CIN layer 1: prev = x0 (ld 40, h<39), K=1521 pad 1536 (24 x 64-wide iters)
    cin_kernel<<<BB, 256, 0, stream>>>(x0row, x0row, 40, FF, W1bf, K1P, K1V, K1P / 64,
                                       conv_b1, nxtrow, res, 1);
    // CIN layer 2: prev = nxt (ld 64, h<64), K=2496 (39 x 64-wide iters)
    cin_kernel<<<BB, 256, 0, stream>>>(x0row, nxtrow, 64, 64, W2bf, K2V, K2V, K2V / 64,
                                       conv_b2, nullptr, res, 0);

    // Deep tower
    gemm_bf16_kernel<<<(BB / 16) * (DEEP1 / 16) / 8, 256, 0, stream>>>(
        flatbf, KD1, W1dT, KD1, deep_b1, y1raw, BB, DEEP1, KD1);
    bnstats_kernel<<<DEEP1, 256, 0, stream>>>(y1raw, DEEP1, mean1, rstd1);
    bnapply_bf16_kernel<<<(BB * KD2) / 256, 256, 0, stream>>>(
        y1raw, mean1, rstd1, bn_g1, bn_b1, y1nbf);

    gemm_bf16_kernel<<<(BB / 16) * (DEEP2 / 16) / 8, 256, 0, stream>>>(
        y1nbf, KD2, W2dT, KD2, deep_b2, y2raw, BB, DEEP2, KD2);
    bnstats_kernel<<<DEEP2, 256, 0, stream>>>(y2raw, DEEP2, mean2, rstd2);
    bnapply_f32_kernel<<<(BB * DEEP2) / 256, 256, 0, stream>>>(
        y2raw, mean2, rstd2, bn_g2, bn_b2, y2);

    final_kernel<<<(BB + 255) / 256, 256, 0, stream>>>(
        flatf, linear_w, linear_b, res, y2, out_w, out_b, out);
}
